// Attention_Fusion_Net_51539607814
// MI455X (gfx1250) — matrix operand order repaired and checked by reading, not compiled, not run on hardware
//
#include <hip/hip_runtime.h>

typedef __attribute__((ext_vector_type(16))) _Float16 v16h;
typedef __attribute__((ext_vector_type(8)))  float    v8f;

#define DEV static __device__ __forceinline__

// ---------------- WMMA fragment helpers (wave32, gfx1250 layouts) ----------------
// A (16x32 f16) from row-major LDS: lanes 0-15 rows M=0..15 hold K 0..15;
// lanes 16-31 hold K 16..31. Two contiguous 8-half groups -> 2x ds_load_b128.
DEV v16h frag_a(const _Float16* t, int ld) {
  int l = threadIdx.x & 31;
  const _Float16* r = t + (l & 15) * ld + ((l >> 4) << 4);
  v16h a;
#pragma unroll
  for (int i = 0; i < 8; ++i) a[i] = r[i];
#pragma unroll
  for (int i = 0; i < 8; ++i) a[8 + i] = r[8 + i];
  return a;
}
// B (32x16 f16) from LDS pre-swizzled into fragment order: lane l's 16 halves are
// contiguous at t + l*16 -> 2x ds_load_b128, no packing movs.
DEV v16h frag_b_sw(const _Float16* t) {
  int l = threadIdx.x & 31;
  const _Float16* p = t + l * 16;
  v16h b;
#pragma unroll
  for (int i = 0; i < 16; ++i) b[i] = p[i];
  return b;
}
// B (32x16 f16) from a TRANSPOSED row-major tile (rows = N/columns of B, ld halves):
// lane l reads 16 contiguous halves of row (l&15) starting at K-offset (l>>4)*16.
DEV v16h frag_bT(const _Float16* t, int ld) {
  int l = threadIdx.x & 31;
  const _Float16* p = t + (l & 15) * ld + ((l >> 4) << 4);
  v16h b;
#pragma unroll
  for (int i = 0; i < 16; ++i) b[i] = p[i];
  return b;
}
DEV v8f wmma_f16(v16h a, v16h b, v8f c) {
  return __builtin_amdgcn_wmma_f32_16x16x32_f16(false, a, false, b, (short)0, c,
                                                false, false);
}
// swizzled B-store index within a 64-row k-chunk holding 2 k-subtiles x NT n-tiles
DEV int bsw_idx(int NT, int kk /*0..63*/, int n) {
  return (((kk >> 5) * NT) + (n >> 4)) * 512 + (((n & 15) | (kk & 16)) << 4) + (kk & 15);
}

// ---------------- generic 1x1-conv-over-positions implicit GEMM ----------------
struct InSpec { const float* p; int nch; int cstr; int bstr; };
struct In4 { InSpec s[4]; };
struct OutSpec { float* p; int cstr; int bstr; };

template <int NT, int MT>
__global__ __launch_bounds__(MT * 2)
void k_conv1x1(In4 in, const float* __restrict__ W, int wmode, int K, int Cout,
               const float* __restrict__ bias, OutSpec o, int relu_flag,
               int Wdim, int Mtot) {
  constexpr int NTN = NT * 16;
  __shared__ _Float16 Bsw[NT * 1024];  // 2 k-subtiles x NT n-tiles, fragment order
  __shared__ _Float16 As[MT * 72];
  const int tid = threadIdx.x;
  const int bdim = MT * 2;
  const int wave = tid >> 5;
  const int nkc = (K + 63) >> 6;
  const int m0 = blockIdx.x * MT;
  v8f acc[NT];
#pragma unroll
  for (int t = 0; t < NT; ++t) acc[t] = {};
  for (int kc = 0; kc < nkc; ++kc) {
    __syncthreads();
    // stage 64-row weight chunk directly in swizzled fragment layout
    for (int idx = tid; idx < 64 * NTN; idx += bdim) {
      int kk = idx / NTN, n = idx - kk * NTN;
      int k = (kc << 6) + kk;
      float v = 0.f;
      if (k < K && n < Cout)
        v = (wmode == 0) ? W[n * K + k] : W[n * K + ((k & 63) << 1) + (k >> 6)];
      Bsw[bsw_idx(NT, kk, n)] = (_Float16)v;
    }
    // stage A chunk (positions x 64 channels), fp32->fp16
    for (int idx = tid; idx < MT * 64; idx += bdim) {
      int r = idx >> 6, kk = idx & 63;
      int k = (kc << 6) + kk;
      int m = m0 + r;
      float v = 0.f;
      if (m < Mtot && k < K) {
        int b = m / Wdim, w = m - b * Wdim;
        int base = 0;
#pragma unroll
        for (int s = 0; s < 4; ++s) {
          int nch = in.s[s].nch;
          if (k < base + nch) {
            v = in.s[s].p[(long)b * in.s[s].bstr + (long)(k - base) * in.s[s].cstr + w];
            break;
          }
          base += nch;
        }
      }
      As[r * 72 + kk] = (_Float16)v;
    }
    __syncthreads();
#pragma unroll
    for (int ks = 0; ks < 2; ++ks) {
      v16h a = frag_a(&As[(wave * 16) * 72 + ks * 32], 72);
#pragma unroll
      for (int t = 0; t < NT; ++t) {
        v16h bf = frag_b_sw(&Bsw[(ks * NT + t) * 512]);
        acc[t] = wmma_f16(a, bf, acc[t]);
      }
    }
  }
  const int l = tid & 31;
  const int col = l & 15;
  const int roff = (l >> 4) << 3;
#pragma unroll
  for (int t = 0; t < NT; ++t) {
    int oc = t * 16 + col;
    if (oc >= Cout) continue;
    float bv = bias ? bias[oc] : 0.f;
#pragma unroll
    for (int i = 0; i < 8; ++i) {
      int m = m0 + wave * 16 + roff + i;
      if (m < Mtot) {
        int b = m / Wdim, w = m - b * Wdim;
        float v = acc[t][i] + bv;
        if (relu_flag) v = fmaxf(v, 0.f);
        o.p[(long)b * o.bstr + (long)oc * o.cstr + w] = v;
      }
    }
  }
}

// ---------------- dedicated im2col convs (ae1/ae2/multiscale/mf_l1) ----------------
// MODE 0: ae1 (K=s*16+c);  1: ae2 (K=ci*3+j over h1);  2: multiscale (K=s*kk+c, padded);
// MODE 3: mf_l1 (K=ci*12+h*3+j over l0buf)
template <int NT, int MODE>
__global__ __launch_bounds__(256)
void k_conv_im(const float* __restrict__ X, const float* __restrict__ W,
               const float* __restrict__ bias, float* __restrict__ outp,
               int K, int Cout, int Wdim, int obstr, int ocstr, int ooff,
               int lkk, int pad, int relu_flag, int Mtot) {
  constexpr int NTN = NT * 16;
  __shared__ _Float16 Bsw[NT * 1024];
  __shared__ _Float16 As[128 * 72];
  const int tid = threadIdx.x;
  const int wave = tid >> 5;
  const int nkc = (K + 63) >> 6;
  const int m0 = blockIdx.x * 128;
  v8f acc[NT];
#pragma unroll
  for (int t = 0; t < NT; ++t) acc[t] = {};
  for (int kc = 0; kc < nkc; ++kc) {
    __syncthreads();
    if (tid == 0 && kc + 1 < nkc) __builtin_prefetch(X + (long)m0, 0, 0); // global_prefetch_b8
    for (int idx = tid; idx < 64 * NTN; idx += 256) {
      int kk = idx / NTN, n = idx - kk * NTN;
      int k = (kc << 6) + kk;
      float v = 0.f;
      if (k < K && n < Cout) v = W[n * K + k];
      Bsw[bsw_idx(NT, kk, n)] = (_Float16)v;
    }
    for (int idx = tid; idx < 128 * 64; idx += 256) {
      int r = idx >> 6, kk = idx & 63;
      int k = (kc << 6) + kk;
      int m = m0 + r;
      float v = 0.f;
      if (m < Mtot && k < K) {
        int b = m / Wdim, pos = m - b * Wdim;
        long idxg = -1;
        if (MODE == 0) {
          int s = k >> 4, c = k & 15;
          idxg = (long)b * 81920 + s * 10240 + pos * 16 + c;
        } else if (MODE == 1) {
          int ci = k / 3, j = k - ci * 3;
          idxg = (long)b * 20480 + ci * 640 + pos * 3 + j;
        } else if (MODE == 2) {
          int s = k >> lkk, c = k & ((1 << lkk) - 1);
          int colx = pos * 16 - pad + c;
          if (colx >= 0 && colx < 10240) idxg = (long)b * 81920 + s * 10240 + colx;
        } else {
          int ci = k / 12, rem = k - ci * 12;
          int h = rem / 3, j = rem - h * 3;
          idxg = (long)b * 20480 + ci * 2560 + h * 640 + pos * 3 + j;
        }
        if (idxg >= 0) v = X[idxg];
      }
      As[r * 72 + kk] = (_Float16)v;
    }
    __syncthreads();
#pragma unroll
    for (int ks = 0; ks < 2; ++ks) {
      v16h a = frag_a(&As[(wave * 16) * 72 + ks * 32], 72);
#pragma unroll
      for (int t = 0; t < NT; ++t) {
        v16h bf = frag_b_sw(&Bsw[(ks * NT + t) * 512]);
        acc[t] = wmma_f16(a, bf, acc[t]);
      }
    }
  }
  const int l = tid & 31;
  const int col = l & 15;
  const int roff = (l >> 4) << 3;
#pragma unroll
  for (int t = 0; t < NT; ++t) {
    int oc = t * 16 + col;
    if (oc >= Cout) continue;
    float bv = bias ? bias[oc] : 0.f;
#pragma unroll
    for (int i = 0; i < 8; ++i) {
      int m = m0 + wave * 16 + roff + i;
      if (m < Mtot) {
        int b = m / Wdim, pos = m - b * Wdim;
        float v = acc[t][i] + bv;
        if (relu_flag) v = fmaxf(v, 0.f);
        outp[(long)b * obstr + (long)oc * ocstr + pos + ooff] = v;
      }
    }
  }
}

// ---------------- axial (W) attention: S=Q^T K, column softmax, out=gamma*(V A)+res ----------------
// LDS: Qs [112][72] @0, Ksw (2x7 swizzled subtiles) @16128, Vs [64][130] @0 (phase3),
//      AsT [112 cols j][136 rows i] @30464 (S stored transposed -> contiguous softmax
//      rows and contiguous phase-3 B-fragments).
template <bool PRE>
__global__ __launch_bounds__(256)
void k_attn(const float* __restrict__ q, const float* __restrict__ k,
            const float* __restrict__ v, const float* __restrict__ Ain,
            const float* __restrict__ gamma, const float* __restrict__ resid,
            int rcstr, int rbstr, float* __restrict__ outp, int ocstr, int obstr,
            float* __restrict__ attn_out) {
#if __has_builtin(__builtin_amdgcn_s_wait_tensorcnt)
  __builtin_amdgcn_s_wait_tensorcnt(0);
#endif
  __shared__ __align__(16) unsigned char smem[60928];
  _Float16* Qs  = (_Float16*)smem;
  _Float16* Ksw = (_Float16*)(smem + 16128);
  _Float16* Vs  = (_Float16*)smem;
  _Float16* AsT = (_Float16*)(smem + 30464);
  const int tid = threadIdx.x;
  const int wave = tid >> 5;
  const int b = blockIdx.x;
  const float* vb = v + (long)b * 6784;
  for (int idx = tid; idx < 112 * 136; idx += 256) AsT[idx] = (_Float16)0.f;
  if (!PRE) {
    const float* qb = q + (long)b * 6784;
    const float* kb = k + (long)b * 6784;
    for (int idx = tid; idx < 112 * 64; idx += 256) {
      int i = idx >> 6, c = idx & 63;
      Qs[i * 72 + c] = (_Float16)((i < 106) ? qb[c * 106 + i] : 0.f);
    }
    for (int idx = tid; idx < 64 * 112; idx += 256) {
      int c = idx / 112, j = idx - c * 112;
      float val = (j < 106) ? kb[c * 106 + j] : 0.f;
      Ksw[(((c >> 5) * 7) + (j >> 4)) * 512 + (((j & 15) | (c & 16)) << 4) + (c & 15)] =
          (_Float16)val;
    }
    __syncthreads();
    for (int tile = wave; tile < 49; tile += 8) {  // 7x7 S tiles
      int mt = tile / 7, nt = tile - mt * 7;
      v8f accS = {};
#pragma unroll
      for (int kc = 0; kc < 2; ++kc) {
        v16h a = frag_a(&Qs[(mt * 16) * 72 + kc * 32], 72);
        v16h bf = frag_b_sw(&Ksw[(kc * 7 + nt) * 512]);
        accS = wmma_f16(a, bf, accS);
      }
      int l = tid & 31, col = nt * 16 + (l & 15), roff = (l >> 4) << 3;
      _Float16* dst = &AsT[col * 136 + mt * 16 + roff];  // 8 contiguous halves
#pragma unroll
      for (int i = 0; i < 8; ++i) dst[i] = (_Float16)accS[i];
    }
    __syncthreads();
    // stage V into region0 (Qs/Ksw dead) + per-column softmax (contiguous in AsT)
    for (int idx = tid; idx < 64 * 128; idx += 256) {
      int c = idx >> 7, i = idx & 127;
      Vs[c * 130 + i] = (_Float16)((i < 106) ? vb[c * 106 + i] : 0.f);
    }
    if (tid < 106) {
      int j = tid;
      _Float16* row = &AsT[j * 136];
      float mx = -1e30f;
      for (int i = 0; i < 106; ++i) mx = fmaxf(mx, (float)row[i]);
      float sum = 0.f;
      for (int i = 0; i < 106; ++i) sum += __expf((float)row[i] - mx);
      float inv = 1.f / sum;
      for (int i = 0; i < 106; ++i) {
        float pv = __expf((float)row[i] - mx) * inv;
        row[i] = (_Float16)pv;
        if (attn_out) attn_out[(long)b * 11236 + i * 106 + j] = pv;
      }
    }
    __syncthreads();
  } else {
    for (int idx = tid; idx < 64 * 128; idx += 256) {
      int c = idx >> 7, i = idx & 127;
      Vs[c * 130 + i] = (_Float16)((i < 106) ? vb[c * 106 + i] : 0.f);
    }
    __syncthreads();  // zero-init of AsT complete before transposed fill
    for (int idx = tid; idx < 106 * 112; idx += 256) {
      int i = idx / 112, j = idx - i * 112;  // coalesced over j
      if (j < 106) AsT[j * 136 + i] = (_Float16)Ain[(long)b * 11236 + i * 106 + j];
    }
    __syncthreads();
  }
  float g = gamma ? gamma[0] : 1.f;
  for (int tile = wave; tile < 28; tile += 8) {  // 4x7 out tiles
    int mt = tile / 7, nt = tile - mt * 7;
    v8f acc = {};
#pragma unroll
    for (int kc = 0; kc < 4; ++kc) {
      v16h a = frag_a(&Vs[(mt * 16) * 130 + kc * 32], 130);
      v16h bf = frag_bT(&AsT[(nt * 16) * 136 + kc * 32], 136);
      acc = wmma_f16(a, bf, acc);
    }
    int l = tid & 31, j = nt * 16 + (l & 15), roff = (l >> 4) << 3;
    if (j < 106) {
#pragma unroll
      for (int i = 0; i < 8; ++i) {
        int c = mt * 16 + roff + i;
        float val = g * acc[i];
        if (resid) val += resid[(long)b * rbstr + (long)c * rcstr + j];
        outp[(long)b * obstr + (long)c * ocstr + j] = val;
      }
    }
  }
}

// ---------------- elementwise / glue kernels ----------------
__global__ __launch_bounds__(256)
void k_pool(const float* __restrict__ in, float* __restrict__ out, int C, long n) {
  long idx = (long)blockIdx.x * 256 + threadIdx.x;
  if (idx >= n) return;
  int u = (int)(idx % 106);
  long r = idx / 106;
  int c = (int)(r % C);
  int b = (int)(r / C);
  const float* p = in + ((long)b * C + c) * 213;
  float v = fmaxf(p[2 * u], p[2 * u + 1]);
  out[idx] = fmaxf(v, 0.f);  // relu(maxpool(.)) == maxpool(relu(.))
}

__global__ __launch_bounds__(256)
void k_mf_l0(const float* __restrict__ in, const float* __restrict__ W,
             const float* __restrict__ bias, float* __restrict__ out) {
  int idx = blockIdx.x * 256 + threadIdx.x;  // B*4*640 positions
  if (idx >= 256 * 2560) return;
  int b = idx / 2560, r = idx - b * 2560;
  const float* p = in + (long)b * 20480 + r;
  float xi[8];
#pragma unroll
  for (int i = 0; i < 8; ++i) xi[i] = p[i * 2560];
  float* o = out + (long)b * 20480 + r;
#pragma unroll
  for (int oc = 0; oc < 8; ++oc) {
    float s = bias[oc];
#pragma unroll
    for (int i = 0; i < 8; ++i) s += W[oc * 8 + i] * xi[i];
    o[oc * 2560] = fmaxf(s, 0.f);
  }
}

// h-branch: out_h = gamma_h*v + x (attn over len-1 axis == 1); Amh/Amh2 stored
// position-major to replicate the reference's reshape reinterpretation.
__global__ __launch_bounds__(256)
void k_outh(const float* __restrict__ vhsc, const float* __restrict__ vhms,
            const float* __restrict__ sc, const float* __restrict__ ms,
            const float* __restrict__ axh_gamma, const float* __restrict__ f23,
            const float* __restrict__ l2c1, const float* __restrict__ g_h1,
            const float* __restrict__ g_h2, float* __restrict__ wout,
            float* __restrict__ Amh, float* __restrict__ Amh2) {
  long idx = (long)blockIdx.x * 256 + threadIdx.x;
  if (idx >= 1736704) return;
  long bw = idx >> 6;
  int c = (int)(idx & 63);
  int b = (int)(bw / 106), w = (int)(bw % 106);
  long cm = (long)b * 6784 + c * 106 + w;
  float vs = vhsc[cm], vm = vhms[cm];
  float gh = axh_gamma[0];
  wout[(long)b * 27136 + c * 424 + w] = gh * vs + sc[cm];          // out_h_SC (h=0)
  wout[(long)b * 27136 + c * 424 + 212 + w] = gh * vm + ms[cm];    // out_h_MS (h=2)
  float lin = f23[0] * vs + f23[1] * vm;                           // xVh
  Amh[idx] = g_h1[0] * lin;                                        // gamma_h1*xVh (scrambled)
  Amh2[idx] = g_h2[0] * (l2c1[0] + l2c1[1]) * lin;                 // method-2 h branch
}

__global__ __launch_bounds__(256)
void k_softmax_mul(const float* __restrict__ pre, const float* __restrict__ ident,
                   float* __restrict__ dst) {
  int idx = blockIdx.x * 256 + threadIdx.x;  // B*W positions
  if (idx >= 27136) return;
  int b = idx / 106, w = idx - b * 106;
  const float* p = pre + (long)b * 6784 + w;
  float mx = -1e30f;
  for (int c = 0; c < 64; ++c) mx = fmaxf(mx, p[c * 106]);
  float sum = 0.f;
  for (int c = 0; c < 64; ++c) sum += __expf(p[c * 106] - mx);
  float inv = 1.f / sum;
  const float* id = ident + (long)b * 6784 + w;
  float* d = dst + (long)b * 6784 + w;
  for (int c = 0; c < 64; ++c) d[c * 106] = id[c * 106] * __expf(p[c * 106] - mx) * inv;
}

// ---------------- launcher ----------------
extern "C" void kernel_launch(void* const* d_in, const int* in_sizes, int n_in,
                              void* d_out, int out_size, void* d_ws, size_t ws_size,
                              hipStream_t stream) {
  (void)in_sizes; (void)n_in; (void)out_size; (void)ws_size;
  const float* X        = (const float*)d_in[0];
  const float* ae_w1    = (const float*)d_in[1];
  const float* ae_b1    = (const float*)d_in[2];
  const float* ae_w2    = (const float*)d_in[3];
  const float* ae_b2    = (const float*)d_in[4];
  const float* mf_w1    = (const float*)d_in[5];
  const float* mf_b1    = (const float*)d_in[6];
  const float* mf_w2    = (const float*)d_in[7];
  const float* mf_b2    = (const float*)d_in[8];
  const float* mf_w3    = (const float*)d_in[9];
  const float* mf_b3    = (const float*)d_in[10];
  const float* mf_w4    = (const float*)d_in[11];
  const float* mf_b4    = (const float*)d_in[12];
  const float* mf_l0_w  = (const float*)d_in[13];
  const float* mf_l0_b  = (const float*)d_in[14];
  const float* mf_l1_w  = (const float*)d_in[15];
  const float* mf_l1_b  = (const float*)d_in[16];
  const float* mf_l2_w  = (const float*)d_in[17];
  const float* mf_l2_b  = (const float*)d_in[18];
  const float* fus0_w   = (const float*)d_in[19];
  // d_in[20..23] = axh_q_w/b, axh_k_w/b : provably dead (softmax over len-1 axis == 1)
  const float* axh_v_w  = (const float*)d_in[24];
  const float* axh_v_b  = (const float*)d_in[25];
  const float* axh_gam  = (const float*)d_in[26];
  const float* axw_q_w  = (const float*)d_in[27];
  const float* axw_q_b  = (const float*)d_in[28];
  const float* axw_k_w  = (const float*)d_in[29];
  const float* axw_k_b  = (const float*)d_in[30];
  const float* axw_v_w  = (const float*)d_in[31];
  const float* axw_v_b  = (const float*)d_in[32];
  const float* axw_gam  = (const float*)d_in[33];
  // f21 (34), f22 (35): dead for same reason
  const float* f23      = (const float*)d_in[36];
  const float* f31      = (const float*)d_in[37];
  const float* f32_     = (const float*)d_in[38];
  const float* f33      = (const float*)d_in[39];
  const float* fus4_w   = (const float*)d_in[40];
  const float* l2c1     = (const float*)d_in[41];
  const float* l2c2     = (const float*)d_in[42];
  const float* l2fus6_w = (const float*)d_in[43];
  const float* fus8_w   = (const float*)d_in[44];
  const float* l4f1_w   = (const float*)d_in[45];
  const float* l4f1_b   = (const float*)d_in[46];
  const float* g_h1     = (const float*)d_in[47];
  const float* g_w1     = (const float*)d_in[48];
  const float* g_h2     = (const float*)d_in[49];
  const float* g_w2     = (const float*)d_in[50];

  float* out = (float*)d_out;
  float* ws  = (float*)d_ws;
  constexpr long SZ = 1736704;    // 256*64*106
  constexpr long AWSZ = 2876416;  // 256*106*106
  float* h1   = ws;               // (B,32,640) then reused as l0 output (B,8,4,640)
  float* t213 = h1 + 5242880;     // (B,64,213) then reused as mf_l1 output (B,32,213)
  float* obuf = t213 + 3489792;   // (B,8,4,640)
  float* sc   = obuf + 5242880;
  float* mfp  = sc + SZ;          // (B,32,106)
  float* ms   = mfp + 868352;
  float* ident= ms + SZ;
  float* vhsc = ident + SZ;
  float* vhms = vhsc + SZ;
  float* qwsc = vhms + SZ;
  float* kwsc = qwsc + SZ;
  float* vwsc = kwsc + SZ;
  float* qwms = vwsc + SZ;
  float* kwms = qwms + SZ;
  float* vwms = kwms + SZ;
  float* xQw  = vwms + SZ;
  float* xKw  = xQw + SZ;
  float* xVw  = xKw + SZ;
  float* Amw  = xVw + SZ;
  float* Amh  = Amw + SZ;
  float* Amh2 = Amh + SZ;
  float* Amw2 = Amh2 + SZ;
  float* preb = Amw2 + SZ;
  float* AwSC = preb + SZ;
  float* AwMS = AwSC + AWSZ;
  float* AwB  = AwMS + AWSZ;
  float* wbase = out + 4 * SZ;    // "weight" output (B,64,4,106)

  // --- encoder / multiscale front-end (implicit GEMM, WMMA f16) ---
  k_conv_im<2, 0><<<1280, 256, 0, stream>>>(X, ae_w1, ae_b1, h1, 128, 32, 640,
                                            20480, 640, 0, 0, 0, 1, 163840);
  k_conv_im<1, 2><<<1280, 256, 0, stream>>>(X, mf_w1, mf_b1, obuf, 128, 8, 640,
                                            20480, 2560, 0, 4, 0, 0, 163840);
  k_conv_im<1, 2><<<1280, 256, 0, stream>>>(X, mf_w2, mf_b2, obuf, 256, 8, 640,
                                            20480, 2560, 640, 5, 8, 0, 163840);
  k_conv_im<1, 2><<<1280, 256, 0, stream>>>(X, mf_w3, mf_b3, obuf, 512, 8, 640,
                                            20480, 2560, 1280, 6, 24, 0, 163840);
  k_conv_im<1, 2><<<1280, 256, 0, stream>>>(X, mf_w4, mf_b4, obuf, 1024, 8, 640,
                                            20480, 2560, 1920, 7, 56, 0, 163840);
  k_conv_im<4, 1><<<426, 256, 0, stream>>>(h1, ae_w2, ae_b2, t213, 96, 64, 213,
                                           13632, 213, 0, 0, 0, 0, 54528);
  k_pool<<<6784, 256, 0, stream>>>(t213, sc, 64, 1736704);
  k_mf_l0<<<2560, 256, 0, stream>>>(obuf, mf_l0_w, mf_l0_b, h1);
  k_conv_im<2, 3><<<426, 256, 0, stream>>>(h1, mf_l1_w, mf_l1_b, t213, 96, 32, 213,
                                           6816, 213, 0, 0, 0, 0, 54528);
  k_pool<<<3392, 256, 0, stream>>>(t213, mfp, 32, 868352);

  auto IS = [](const float* p, int nch, int cstr, int bstr) {
    InSpec s; s.p = p; s.nch = nch; s.cstr = cstr; s.bstr = bstr; return s;
  };
  InSpec Z = IS(nullptr, 0, 0, 0);
  auto conv64 = [&](InSpec a, InSpec b2, const float* W, int wmode, int K,
                    const float* bias, float* op, int ocstr, int obstr, int relu) {
    In4 in; in.s[0] = a; in.s[1] = b2; in.s[2] = Z; in.s[3] = Z;
    OutSpec o; o.p = op; o.cstr = ocstr; o.bstr = obstr;
    k_conv1x1<4, 128><<<212, 256, 0, stream>>>(in, W, wmode, K, 64, bias, o, relu,
                                               106, 27136);
  };
  InSpec scS = IS(sc, 64, 106, 6784), msS = IS(ms, 64, 106, 6784);

  conv64(IS(mfp, 32, 106, 3392), Z, mf_l2_w, 0, 32, mf_l2_b, ms, 106, 6784, 1);
  conv64(scS, msS, fus0_w, 0, 128, nullptr, ident, 106, 6784, 1);         // identity
  conv64(scS, msS, l4f1_w, 1, 128, l4f1_b, out + 3 * SZ, 106, 6784, 0);   // out4
  conv64(scS, Z, axh_v_w, 0, 64, axh_v_b, vhsc, 106, 6784, 0);
  conv64(msS, Z, axh_v_w, 0, 64, axh_v_b, vhms, 106, 6784, 0);
  conv64(scS, Z, axw_q_w, 0, 64, axw_q_b, qwsc, 106, 6784, 0);
  conv64(scS, Z, axw_k_w, 0, 64, axw_k_b, kwsc, 106, 6784, 0);
  conv64(scS, Z, axw_v_w, 0, 64, axw_v_b, vwsc, 106, 6784, 0);
  conv64(msS, Z, axw_q_w, 0, 64, axw_q_b, qwms, 106, 6784, 0);
  conv64(msS, Z, axw_k_w, 0, 64, axw_k_b, kwms, 106, 6784, 0);
  conv64(msS, Z, axw_v_w, 0, 64, axw_v_b, vwms, 106, 6784, 0);

  // axial-W attention for both branches; out_w written straight into `weight`
  k_attn<false><<<256, 256, 0, stream>>>(qwsc, kwsc, vwsc, nullptr, axw_gam,
                                         sc, 106, 6784, wbase + 106, 424, 27136, AwSC);
  k_attn<false><<<256, 256, 0, stream>>>(qwms, kwms, vwms, nullptr, axw_gam,
                                         ms, 106, 6784, wbase + 318, 424, 27136, AwMS);
  k_outh<<<6784, 256, 0, stream>>>(vhsc, vhms, sc, ms, axh_gam, f23, l2c1,
                                   g_h1, g_h2, wbase, Amh, Amh2);

  // --- method 1 ---
  conv64(IS(qwsc, 64, 106, 6784), IS(qwms, 64, 106, 6784), f31, 0, 128, nullptr, xQw, 106, 6784, 0);
  conv64(IS(kwsc, 64, 106, 6784), IS(kwms, 64, 106, 6784), f32_, 0, 128, nullptr, xKw, 106, 6784, 0);
  conv64(IS(vwsc, 64, 106, 6784), IS(vwms, 64, 106, 6784), f33, 0, 128, nullptr, xVw, 106, 6784, 0);
  k_attn<false><<<256, 256, 0, stream>>>(xQw, xKw, xVw, nullptr, g_w1,
                                         nullptr, 0, 0, Amw, 106, 6784, nullptr);
  conv64(IS(Amw, 64, 106, 6784), IS(Amh, 64, 106, 6784), fus4_w, 0, 128, nullptr,
         preb, 106, 6784, 0);
  k_softmax_mul<<<106, 256, 0, stream>>>(preb, ident, out);                // out1

  // --- method 2 ---
  {
    In4 in; in.s[0] = IS(AwSC, 106, 106, 11236); in.s[1] = IS(AwMS, 106, 106, 11236);
    in.s[2] = Z; in.s[3] = Z;
    OutSpec o; o.p = AwB; o.cstr = 106; o.bstr = 11236;
    k_conv1x1<7, 64><<<424, 128, 0, stream>>>(in, l2c2, 0, 212, 106, nullptr, o, 0,
                                              106, 27136);
  }
  k_attn<true><<<256, 256, 0, stream>>>(nullptr, nullptr, xVw, AwB, g_w2,
                                        nullptr, 0, 0, Amw2, 106, 6784, nullptr);
  conv64(IS(Amw2, 64, 106, 6784), IS(Amh2, 64, 106, 6784), l2fus6_w, 0, 128,
         nullptr, preb, 106, 6784, 0);
  k_softmax_mul<<<106, 256, 0, stream>>>(preb, ident, out + SZ);           // out2

  // --- method 3: fuse the four attention outputs already living in `weight` ---
  {
    In4 in;
    for (int h = 0; h < 4; ++h) in.s[h] = IS(wbase + h * 106, 64, 424, 27136);
    OutSpec o; o.p = preb; o.cstr = 106; o.bstr = 6784;
    k_conv1x1<4, 128><<<212, 256, 0, stream>>>(in, fus8_w, 0, 256, 64, nullptr, o, 0,
                                               106, 27136);
  }
  k_softmax_mul<<<106, 256, 0, stream>>>(preb, ident, out + 2 * SZ);       // out3
}
